// DeepSeekV2MoE_26087631356410
// MI455X (gfx1250) — matrix operand order repaired and checked by reading, not compiled, not run on hardware
//
#include <hip/hip_runtime.h>
#include <hip/hip_bf16.h>

#define H 2048
#define F 1408
#define T 2048
#define E 8
#define TOPK 2

typedef __bf16 v16bf __attribute__((ext_vector_type(16)));
typedef float  v8f   __attribute__((ext_vector_type(8)));
typedef int    v4i_t __attribute__((ext_vector_type(4)));

union FragU { uint4 q[2]; v16bf v; };

#ifndef __has_builtin
#define __has_builtin(x) 0
#endif

#if __has_builtin(__builtin_amdgcn_global_load_async_to_lds_b128)
#define HAVE_ASYNC_LDS 1
typedef __attribute__((address_space(1))) v4i_t* gptr_v4i;
typedef __attribute__((address_space(3))) v4i_t* lptr_v4i;
#else
#define HAVE_ASYNC_LDS 0
#endif

#if __has_builtin(__builtin_amdgcn_s_wait_asynccnt)
#define WAIT_ASYNC(n) __builtin_amdgcn_s_wait_asynccnt(n)
#else
#define WAIT_ASYNC(n) asm volatile("s_wait_asynccnt %0" :: "i"(n) : "memory")
#endif

// async 16-byte copy global -> LDS (ASYNCcnt-tracked); sync fallback
__device__ __forceinline__ void async_cp16(const unsigned short* g, unsigned short* l) {
#if HAVE_ASYNC_LDS
    __builtin_amdgcn_global_load_async_to_lds_b128(
        (gptr_v4i)(void*)g, (lptr_v4i)(void*)l, 0, 0);
#else
    *(uint4*)l = *(const uint4*)g;
#endif
}

__device__ __forceinline__ unsigned short f2bf(float f) {
    unsigned u = __float_as_uint(f);
    unsigned r = u + 0x7FFFu + ((u >> 16) & 1u);
    return (unsigned short)(r >> 16);
}

__device__ __forceinline__ float fast_silu(float g) {
    return g * __builtin_amdgcn_rcpf(1.0f + __expf(-g));
}

// ---------------- Kernel 0: fp32 -> bf16 convert (vector x4) ----------------
__global__ __launch_bounds__(256) void cvt_bf16_kernel(
    const float* __restrict__ x, unsigned short* __restrict__ xb)
{
    int i = blockIdx.x * blockDim.x + threadIdx.x;
    float4 v = ((const float4*)x)[i];
    uint2 o;
    o.x = (unsigned)f2bf(v.x) | ((unsigned)f2bf(v.y) << 16);
    o.y = (unsigned)f2bf(v.z) | ((unsigned)f2bf(v.w) << 16);
    ((uint2*)xb)[i] = o;
}

// ------- Kernel 0b: per-expert fp32 [K][N] -> bf16 transposed [N][K] --------
__global__ __launch_bounds__(256) void transpose_bf16_kernel(
    const float* __restrict__ in, unsigned short* __restrict__ outp,
    int K, int N)
{
    __shared__ unsigned short tile[32][33];
    const float* src = in + (size_t)blockIdx.z * K * N;
    unsigned short* dst = outp + (size_t)blockIdx.z * K * N;
    const int k0 = blockIdx.y * 32;
    const int n0 = blockIdx.x * 32;
    const int tx = threadIdx.x & 31;
    const int ty = threadIdx.x >> 5;
#pragma unroll
    for (int i = 0; i < 32; i += 8)
        tile[ty + i][tx] = f2bf(src[(size_t)(k0 + ty + i) * N + n0 + tx]);
    __syncthreads();
#pragma unroll
    for (int i = 0; i < 32; i += 8)
        dst[(size_t)(n0 + ty + i) * K + k0 + tx] = tile[tx][ty + i];
}

// ---------------- Kernel 1: router (softmax + top2 + compaction) ------------
__global__ __launch_bounds__(256) void router_kernel(
    const float* __restrict__ x, const float* __restrict__ gate_w,
    int* __restrict__ cnt, int* __restrict__ idxp, float* __restrict__ wtp)
{
    const int wave = threadIdx.x >> 5;
    const int lane = threadIdx.x & 31;
    const int t = blockIdx.x * 8 + wave;

    float acc[E];
#pragma unroll
    for (int e = 0; e < E; ++e) acc[e] = 0.0f;

    const float* xr = x + (size_t)t * H;
    for (int h = lane; h < H; h += 32) {
        float xv = xr[h];
        const float* g = gate_w + h * E;
#pragma unroll
        for (int e = 0; e < E; ++e) acc[e] = fmaf(xv, g[e], acc[e]);
    }
#pragma unroll
    for (int off = 16; off > 0; off >>= 1) {
#pragma unroll
        for (int e = 0; e < E; ++e) acc[e] += __shfl_xor(acc[e], off, 32);
    }
    if (lane == 0) {
        float mx = acc[0];
#pragma unroll
        for (int e = 1; e < E; ++e) mx = fmaxf(mx, acc[e]);
        float p[E];
#pragma unroll
        for (int e = 0; e < E; ++e) p[e] = __expf(acc[e] - mx);
        int e0 = 0;
#pragma unroll
        for (int e = 1; e < E; ++e) if (p[e] > p[e0]) e0 = e;
        int e1 = (e0 == 0) ? 1 : 0;
#pragma unroll
        for (int e = 0; e < E; ++e) if (e != e0 && p[e] > p[e1]) e1 = e;
        float w0 = p[e0] * __builtin_amdgcn_rcpf(p[e0] + p[e1]);
        float w1 = 1.0f - w0;
        int s0 = atomicAdd(&cnt[e0], 1);
        idxp[e0 * T + s0] = t; wtp[e0 * T + s0] = w0;
        int s1 = atomicAdd(&cnt[e1], 1);
        idxp[e1 * T + s1] = t; wtp[e1 * T + s1] = w1;
    }
}

// ---------------- Kernel 2: exclusive prefix of counts ----------------------
__global__ void prefix_kernel(const int* __restrict__ cnt, int* __restrict__ basep)
{
    int s = 0;
    for (int e = 0; e < E; ++e) { basep[e] = s; s += cnt[e]; }
    basep[E] = s;
}

#define LDS_K 40   // 32 + 8 pad halfwords

// ---------------- Kernel 3: gate/up GEMM + fused SwiGLU ---------------------
// 64x64 block tile, K-step 32, double-buffered LDS fed by async copies.
__global__ __launch_bounds__(256) void moe_gateup_kernel(
    const unsigned short* __restrict__ xb,
    const unsigned short* __restrict__ wgb,   // [E][F][H] bf16 (transposed)
    const unsigned short* __restrict__ wub,   // [E][F][H] bf16 (transposed)
    const int* __restrict__ cnt, const int* __restrict__ basep,
    const int* __restrict__ idxp,
    unsigned short* __restrict__ Aws)
{
    const int e = blockIdx.z;
    const int count = cnt[e];
    const int m_base = blockIdx.y * 64;
    if (m_base >= count) return;
    const int f0 = blockIdx.x * 64;

    const int tid = threadIdx.x;
    const int lane = tid & 31;
    const int wave = tid >> 5;
    const int wm = wave >> 2;
    const int wn = wave & 3;

    __shared__ int tokens[64];
    __shared__ unsigned short As[2][64 * LDS_K];
    __shared__ unsigned short Bg[2][64 * LDS_K];
    __shared__ unsigned short Bu[2][64 * LDS_K];

    if (tid < 64) {
        int slot = m_base + tid;
        int sc = (slot < count) ? slot : (count - 1);
        tokens[tid] = idxp[e * T + sc];
    }
    __syncthreads();

    const unsigned short* Wg = wgb + ((size_t)e * F + f0) * H;
    const unsigned short* Wu = wub + ((size_t)e * F + f0) * H;

    v8f cg[2] = {{}, {}};
    v8f cu[2] = {{}, {}};

    const int srow = tid >> 2;
    const int sk8  = (tid & 3) * 8;
    const int soff = srow * LDS_K + sk8;

    const unsigned short* aSrc = xb + (size_t)tokens[srow] * H + sk8;
    const unsigned short* gSrc = Wg + (size_t)srow * H + sk8;
    const unsigned short* uSrc = Wu + (size_t)srow * H + sk8;

    // prologue: stage step 0 into buffer 0
    async_cp16(aSrc, &As[0][soff]);
    async_cp16(gSrc, &Bg[0][soff]);
    async_cp16(uSrc, &Bu[0][soff]);

    const int nsteps = H / 32;
    for (int i = 0; i < nsteps; ++i) {
        const int b = i & 1;
        if (i + 1 < nsteps) {
            const int k1 = (i + 1) * 32;
            async_cp16(aSrc + k1, &As[1 - b][soff]);
            async_cp16(gSrc + k1, &Bg[1 - b][soff]);
            async_cp16(uSrc + k1, &Bu[1 - b][soff]);
            WAIT_ASYNC(3);   // step i's 3 copies done; step i+1 in flight
        } else {
            WAIT_ASYNC(0);
        }
        __syncthreads();     // all waves' step-i tiles visible

        FragU a0, a1, bg, bu;
        const int ar = wm * 32 + (lane & 15);
        const int ac0 = (lane < 16) ? 0 : 8;
        a0.q[0] = *(const uint4*)&As[b][ar * LDS_K + ac0];
        a0.q[1] = *(const uint4*)&As[b][ar * LDS_K + ac0 + 16];
        a1.q[0] = *(const uint4*)&As[b][(ar + 16) * LDS_K + ac0];
        a1.q[1] = *(const uint4*)&As[b][(ar + 16) * LDS_K + ac0 + 16];
        const int bc = wn * 16 + (lane & 15);
        const int bkb = (lane < 16) ? 0 : 8;
        bg.q[0] = *(const uint4*)&Bg[b][bc * LDS_K + bkb];
        bg.q[1] = *(const uint4*)&Bg[b][bc * LDS_K + bkb + 16];
        bu.q[0] = *(const uint4*)&Bu[b][bc * LDS_K + bkb];
        bu.q[1] = *(const uint4*)&Bu[b][bc * LDS_K + bkb + 16];

        cg[0] = __builtin_amdgcn_wmma_f32_16x16x32_bf16(false, a0.v, false, bg.v,
                                                        (short)0, cg[0], false, false);
        cg[1] = __builtin_amdgcn_wmma_f32_16x16x32_bf16(false, a1.v, false, bg.v,
                                                        (short)0, cg[1], false, false);
        cu[0] = __builtin_amdgcn_wmma_f32_16x16x32_bf16(false, a0.v, false, bu.v,
                                                        (short)0, cu[0], false, false);
        cu[1] = __builtin_amdgcn_wmma_f32_16x16x32_bf16(false, a1.v, false, bu.v,
                                                        (short)0, cu[1], false, false);
        __syncthreads();     // protect buffer b before it is re-staged
    }

    const int mybase = basep[e];
    const int n = f0 + wn * 16 + (lane & 15);
    const int mo = (lane < 16) ? 0 : 8;
#pragma unroll
    for (int r = 0; r < 2; ++r) {
#pragma unroll
        for (int i = 0; i < 8; ++i) {
            int ml = wm * 32 + r * 16 + mo + i;
            int slot = m_base + ml;
            if (slot < count) {
                float av = fast_silu(cg[r][i]) * cu[r][i];
                Aws[(size_t)(mybase + slot) * F + n] = f2bf(av);
            }
        }
    }
}

// ---------------- Kernel 4: down GEMM + weighted atomic scatter -------------
__global__ __launch_bounds__(256) void moe_down_kernel(
    const unsigned short* __restrict__ Aws,
    const unsigned short* __restrict__ wdb,   // [E][H][F] bf16 (transposed)
    const int* __restrict__ cnt, const int* __restrict__ basep,
    const int* __restrict__ idxp, const float* __restrict__ wtp,
    float* __restrict__ out)
{
    const int e = blockIdx.z;
    const int count = cnt[e];
    const int m_base = blockIdx.y * 64;
    if (m_base >= count) return;
    const int h0 = blockIdx.x * 64;

    const int tid = threadIdx.x;
    const int lane = tid & 31;
    const int wave = tid >> 5;
    const int wm = wave >> 2;
    const int wn = wave & 3;

    __shared__ int tokens[64];
    __shared__ float wrow[64];
    __shared__ unsigned short As[2][64 * LDS_K];
    __shared__ unsigned short Bd[2][64 * LDS_K];

    const int rowbase = basep[e];

    if (tid < 64) {
        int slot = m_base + tid;
        int sc = (slot < count) ? slot : (count - 1);
        tokens[tid] = idxp[e * T + sc];
        wrow[tid] = (slot < count) ? wtp[e * T + slot] : 0.0f;
    }
    __syncthreads();

    const unsigned short* Wd = wdb + ((size_t)e * H + h0) * F;

    v8f c[2] = {{}, {}};

    const int srow = tid >> 2;
    const int sk8  = (tid & 3) * 8;
    const int soff = srow * LDS_K + sk8;

    int aslot = m_base + srow;
    if (aslot >= count) aslot = count - 1;
    const unsigned short* aSrc = Aws + (size_t)(rowbase + aslot) * F + sk8;
    const unsigned short* dSrc = Wd + (size_t)srow * F + sk8;

    async_cp16(aSrc, &As[0][soff]);
    async_cp16(dSrc, &Bd[0][soff]);

    const int nsteps = F / 32;
    for (int i = 0; i < nsteps; ++i) {
        const int b = i & 1;
        if (i + 1 < nsteps) {
            const int k1 = (i + 1) * 32;
            async_cp16(aSrc + k1, &As[1 - b][soff]);
            async_cp16(dSrc + k1, &Bd[1 - b][soff]);
            WAIT_ASYNC(2);
        } else {
            WAIT_ASYNC(0);
        }
        __syncthreads();

        FragU a0, a1, bfr;
        const int ar = wm * 32 + (lane & 15);
        const int ac0 = (lane < 16) ? 0 : 8;
        a0.q[0] = *(const uint4*)&As[b][ar * LDS_K + ac0];
        a0.q[1] = *(const uint4*)&As[b][ar * LDS_K + ac0 + 16];
        a1.q[0] = *(const uint4*)&As[b][(ar + 16) * LDS_K + ac0];
        a1.q[1] = *(const uint4*)&As[b][(ar + 16) * LDS_K + ac0 + 16];
        const int bc = wn * 16 + (lane & 15);
        const int bkb = (lane < 16) ? 0 : 8;
        bfr.q[0] = *(const uint4*)&Bd[b][bc * LDS_K + bkb];
        bfr.q[1] = *(const uint4*)&Bd[b][bc * LDS_K + bkb + 16];

        c[0] = __builtin_amdgcn_wmma_f32_16x16x32_bf16(false, a0.v, false, bfr.v,
                                                       (short)0, c[0], false, false);
        c[1] = __builtin_amdgcn_wmma_f32_16x16x32_bf16(false, a1.v, false, bfr.v,
                                                       (short)0, c[1], false, false);
        __syncthreads();
    }

    const int n = h0 + wn * 16 + (lane & 15);
    const int mo = (lane < 16) ? 0 : 8;
#pragma unroll
    for (int r = 0; r < 2; ++r) {
#pragma unroll
        for (int i = 0; i < 8; ++i) {
            int ml = wm * 32 + r * 16 + mo + i;
            int slot = m_base + ml;
            if (slot < count) {
                float v = c[r][i] * wrow[ml];
                atomicAdd(&out[(size_t)tokens[ml] * H + n], v);
            }
        }
    }
}

// ---------------------------------------------------------------------------
extern "C" void kernel_launch(void* const* d_in, const int* in_sizes, int n_in,
                              void* d_out, int out_size, void* d_ws, size_t ws_size,
                              hipStream_t stream) {
    (void)in_sizes; (void)n_in; (void)out_size; (void)ws_size;
    const float* x      = (const float*)d_in[0];
    const float* gate_w = (const float*)d_in[1];
    const float* w_gate = (const float*)d_in[2];
    const float* w_up   = (const float*)d_in[3];
    const float* w_down = (const float*)d_in[4];
    float* out = (float*)d_out;

    char* ws = (char*)d_ws;
    size_t off = 0;
    auto carve = [&](size_t bytes) -> void* {
        void* p = ws + off;
        off = (off + bytes + 255) & ~(size_t)255;
        return p;
    };
    unsigned short* xb  = (unsigned short*)carve((size_t)T * H * 2);
    unsigned short* Aws = (unsigned short*)carve((size_t)T * TOPK * F * 2);
    unsigned short* wgb = (unsigned short*)carve((size_t)E * H * F * 2);
    unsigned short* wub = (unsigned short*)carve((size_t)E * H * F * 2);
    unsigned short* wdb = (unsigned short*)carve((size_t)E * H * F * 2);
    int*   cnt   = (int*)carve(E * sizeof(int));
    int*   basep = (int*)carve((E + 1) * sizeof(int));
    int*   idxp  = (int*)carve((size_t)E * T * sizeof(int));
    float* wtp   = (float*)carve((size_t)E * T * sizeof(float));

    (void)hipMemsetAsync(cnt, 0, E * sizeof(int), stream);
    (void)hipMemsetAsync(out, 0, (size_t)T * H * sizeof(float), stream);

    cvt_bf16_kernel<<<(T * H / 4) / 256, 256, 0, stream>>>(x, xb);

    dim3 tg_gu(F / 32, H / 32, E);
    transpose_bf16_kernel<<<tg_gu, 256, 0, stream>>>(w_gate, wgb, H, F);
    transpose_bf16_kernel<<<tg_gu, 256, 0, stream>>>(w_up,   wub, H, F);
    dim3 tg_d(H / 32, F / 32, E);
    transpose_bf16_kernel<<<tg_d, 256, 0, stream>>>(w_down, wdb, F, H);

    router_kernel<<<T / 8, 256, 0, stream>>>(x, gate_w, cnt, idxp, wtp);
    prefix_kernel<<<1, 1, 0, stream>>>(cnt, basep);

    dim3 g2(F / 64, T / 64, E);
    moe_gateup_kernel<<<g2, 256, 0, stream>>>(xb, wgb, wub, cnt, basep, idxp, Aws);

    dim3 g3(H / 64, T / 64, E);
    moe_down_kernel<<<g3, 256, 0, stream>>>(Aws, wdb, cnt, basep, idxp, wtp, out);
}
